// CPCFGs_30210799960287
// MI455X (gfx1250) — compile-verified, matrix operand order checked by reading, not run on hardware
//
#include <hip/hip_runtime.h>

typedef _Float16 h16;
typedef __attribute__((ext_vector_type(16))) _Float16 v16h;
typedef __attribute__((ext_vector_type(8)))  _Float16 v8h;
typedef __attribute__((ext_vector_type(8)))  float    v8f;

#define B_  32
#define N_  24
#define V_  10000
#define NT_ 15
#define T_  30
#define S_  45
#define SS_ 2025
#define SD_ 256
#define ZD_ 64
#define HD_ 512
#define WD_ 512
#define NEGV -1.0e9f

// CDNA5 async global->LDS (per-lane gather, tracked by ASYNCcnt)
#define ASYNC_LD_B32(ldsAddr, gptr) \
  asm volatile("global_load_async_to_lds_b32 %0, %1, off" :: "v"(ldsAddr), "v"(gptr) : "memory")
#define WAIT_ASYNC() asm volatile("s_wait_asynccnt 0x0" ::: "memory")

__device__ __forceinline__ float sigm(float x) { return 1.f / (1.f + expf(-x)); }

__device__ __forceinline__ v16h load_frag16(const h16* p) {
  v8h lo = *(const v8h*)(p);
  v8h hi = *(const v8h*)(p + 16);
  v16h r;
#pragma unroll
  for (int i = 0; i < 8; ++i) { r[i] = lo[i]; r[8 + i] = hi[i]; }
  return r;
}

__device__ __forceinline__ v8f wmma32(const v16h& a, const v16h& b, const v8f& c) {
  return __builtin_amdgcn_wmma_f32_16x16x32_f16(false, a, false, b, (short)0, c, false, false);
}

// C = act(A * Bt^T + bias). A: MxK f16 row-major (rows padded to 32). Bt: NxK f16
// row-major (rows padded to 32). K % 32 == 0 (zero padded).
// One wave computes a 32x32 macro-tile: 4 accumulators, 4 WMMAs per K-step.
__global__ void wmma_gemm(const h16* __restrict__ A, int lda, long sA,
                          const h16* __restrict__ Bt, int ldb, long sB,
                          const float* __restrict__ bias,
                          float* __restrict__ Cf, int ldcf, long sCf,
                          h16* __restrict__ Ch, int ldch, long sCh,
                          int M, int N, int K, int act)
{
  const int bz = blockIdx.z;
  A  += (long)bz * sA;
  Bt += (long)bz * sB;
  if (Cf) Cf += (long)bz * sCf;
  if (Ch) Ch += (long)bz * sCh;
  const int lane = threadIdx.x & 31;
  const int half = lane >> 4;
  const int l15  = lane & 15;
  const int mb = blockIdx.y * 32, nb = blockIdx.x * 32;
  const h16* a0 = A  + (long)(mb + l15) * lda + half * 8;
  const h16* a1 = a0 + (long)16 * lda;
  const h16* b0 = Bt + (long)(nb + l15) * ldb + half * 8;
  const h16* b1 = b0 + (long)16 * ldb;
  v8f c00 = {}, c01 = {}, c10 = {}, c11 = {};
  for (int k = 0; k < K; k += 32) {
    __builtin_prefetch(a0 + k + 128, 0, 0);
    __builtin_prefetch(a1 + k + 128, 0, 0);
    __builtin_prefetch(b0 + k + 128, 0, 0);
    __builtin_prefetch(b1 + k + 128, 0, 0);
    v16h fa0 = load_frag16(a0 + k);
    v16h fa1 = load_frag16(a1 + k);
    v16h fb0 = load_frag16(b0 + k);
    v16h fb1 = load_frag16(b1 + k);
    c00 = wmma32(fa0, fb0, c00);
    c01 = wmma32(fa0, fb1, c01);
    c10 = wmma32(fa1, fb0, c10);
    c11 = wmma32(fa1, fb1, c11);
  }
  auto store_tile = [&](const v8f& acc, int mbase, int nbase) {
    int n = nbase + l15;
    int m0 = mbase + half * 8;
    float bv = (bias != nullptr && n < N) ? bias[n] : 0.f;
#pragma unroll
    for (int r = 0; r < 8; ++r) {
      int m = m0 + r;
      if (m < M && n < N) {
        float v = acc[r] + bv;
        if (act == 1) v = fmaxf(v, 0.f);
        if (Cf) Cf[(long)m * ldcf + n] = v;
        if (Ch) Ch[(long)m * ldch + n] = (h16)v;
      }
    }
  };
  store_tile(c00, mb, nb);
  store_tile(c01, mb, nb + 16);
  store_tile(c10, mb + 16, nb);
  store_tile(c11, mb + 16, nb + 16);
}

// chart[i, i+w-1, b, a] = log( F_b(i,:) . Rexp_b(a,:) ) + cm[i,b] + rmax[b,a]
// 2 M-subtiles sharing one B fragment (N = 16 covers NT_ = 15).
__global__ void inside_gemm(const h16* __restrict__ F, const h16* __restrict__ Rexp,
                            const float* __restrict__ cm, const float* __restrict__ rmaxv,
                            float* __restrict__ chart, int w, int ni)
{
  const int b = blockIdx.y;
  const int lane = threadIdx.x & 31;
  const int half = lane >> 4;
  const int l15  = lane & 15;
  const h16* a0 = F    + ((long)b * 32 + l15) * 2048 + half * 8;
  const h16* a1 = a0 + (long)16 * 2048;
  const h16* bp = Rexp + ((long)b * 16 + l15) * 2048 + half * 8;
  v8f c0 = {}, c1 = {};
  for (int k = 0; k < 2048; k += 32) {
    __builtin_prefetch(a0 + k + 128, 0, 0);
    __builtin_prefetch(bp + k + 128, 0, 0);
    v16h fa0 = load_frag16(a0 + k);
    v16h fa1 = load_frag16(a1 + k);
    v16h fb  = load_frag16(bp + k);
    c0 = wmma32(fa0, fb, c0);
    c1 = wmma32(fa1, fb, c1);
  }
  const int a_ = l15;
  auto store_tile = [&](const v8f& acc, int mbase) {
    int m0 = mbase + half * 8;
#pragma unroll
    for (int r = 0; r < 8; ++r) {
      int i = m0 + r;
      if (i < ni && a_ < NT_) {
        int j = i + w - 1;
        chart[(((long)i * N_ + j) * B_ + b) * S_ + a_] =
            logf(fmaxf(acc[r], 1e-30f)) + cm[i * B_ + b] + rmaxv[b * NT_ + a_];
      }
    }
  };
  store_tile(c0, 0);
  store_tile(c1, 16);
}

// cm[i,b] = max_u(lmax_u + rmax_u); F[b,i,xy] = sum_u exp(l-lm_u)*exp(r-(cm-lm_u))
__global__ void logz_combine(const float* __restrict__ chart, h16* __restrict__ F,
                             float* __restrict__ cm, int w)
{
  const int U = w - 1;
  const int bid = blockIdx.x;
  const int i = bid / B_, b = bid % B_;
  const int tid = threadIdx.x;
  __shared__ float sl[23 * S_], sr[23 * S_], slm[23], srm[23], scm;
  const int tot = U * S_;
  for (int idx = tid; idx < tot; idx += 256) {
    int u = idx / S_, x = idx % S_;
    const float* gl = chart + (((long)i * N_ + (i + u)) * B_ + b) * S_ + x;
    const float* gr = chart + (((long)(i + u + 1) * N_ + (i + w - 1)) * B_ + b) * S_ + x;
    ASYNC_LD_B32((unsigned)(size_t)(&sl[idx]), gl);
    ASYNC_LD_B32((unsigned)(size_t)(&sr[idx]), gr);
  }
  WAIT_ASYNC();
  __syncthreads();
  if (tid < U) {
    float m1 = -3.0e38f, m2 = -3.0e38f;
    for (int x = 0; x < S_; ++x) { m1 = fmaxf(m1, sl[tid * S_ + x]); m2 = fmaxf(m2, sr[tid * S_ + x]); }
    slm[tid] = m1; srm[tid] = m2;
  }
  __syncthreads();
  if (tid == 0) {
    float c = -3.0e38f;
    for (int u = 0; u < U; ++u) c = fmaxf(c, slm[u] + srm[u]);
    scm = c;
    cm[i * B_ + b] = c;
  }
  __syncthreads();
  for (int idx = tid; idx < tot; idx += 256) {
    int u = idx / S_;
    sl[idx] = expf(sl[idx] - slm[u]);
    sr[idx] = expf(sr[idx] - (scm - slm[u]));
  }
  __syncthreads();
  for (int xy = tid; xy < SS_; xy += 256) {
    int x = xy / S_, y = xy % S_;
    float s = 0.f;
    for (int u = 0; u < U; ++u) s += sl[u * S_ + x] * sr[u * S_ + y];
    F[((long)b * 32 + i) * 2048 + xy] = (h16)s;
  }
}

// G[x,y] = max_u(l[u,x] + r[u,y]) with arg-u
__global__ void vit_combine(const float* __restrict__ chart, float* __restrict__ G,
                            unsigned char* __restrict__ argU, int w)
{
  const int U = w - 1;
  const int bid = blockIdx.x;
  const int i = bid / B_, b = bid % B_;
  const int tid = threadIdx.x;
  __shared__ float sl[23 * S_], sr[23 * S_];
  const int tot = U * S_;
  for (int idx = tid; idx < tot; idx += 256) {
    int u = idx / S_, x = idx % S_;
    const float* gl = chart + (((long)i * N_ + (i + u)) * B_ + b) * S_ + x;
    const float* gr = chart + (((long)(i + u + 1) * N_ + (i + w - 1)) * B_ + b) * S_ + x;
    ASYNC_LD_B32((unsigned)(size_t)(&sl[idx]), gl);
    ASYNC_LD_B32((unsigned)(size_t)(&sr[idx]), gr);
  }
  WAIT_ASYNC();
  __syncthreads();
  for (int xy = tid; xy < SS_; xy += 256) {
    int x = xy / S_, y = xy % S_;
    float m = -3.0e38f; int bu = 0;
    for (int u = 0; u < U; ++u) {
      float v = sl[u * S_ + x] + sr[u * S_ + y];
      if (v > m) { m = v; bu = u; }
    }
    G[(long)bid * SS_ + xy] = m;
    argU[(long)bid * SS_ + xy] = (unsigned char)bu;
  }
}

// chart[i,j,b,a] = max_xy(rules[b,a,xy] + G[xy]); bp packs (x,y,u)
__global__ void vit_reduce(const float* __restrict__ G, const unsigned char* __restrict__ argU,
                           const float* __restrict__ rules, float* __restrict__ chart,
                           int* __restrict__ bp, int w)
{
  const int bid = blockIdx.x;
  const int i = bid / B_, b = bid % B_;
  const int j = i + w - 1;
  const int tid = threadIdx.x;
  __shared__ float sg[SS_];
  __shared__ float rv[256];
  __shared__ int   ri[256];
  for (int xy = tid; xy < SS_; xy += 256) {
    const float* gp = G + (long)bid * SS_ + xy;
    ASYNC_LD_B32((unsigned)(size_t)(&sg[xy]), gp);
  }
  WAIT_ASYNC();
  __syncthreads();
  for (int a = 0; a < NT_; ++a) {
    const float* rr = rules + ((long)b * NT_ + a) * SS_;
    float m = -3.0e38f; int mi = 0;
    for (int xy = tid; xy < SS_; xy += 256) {
      float v = rr[xy] + sg[xy];
      if (v > m) { m = v; mi = xy; }
    }
    rv[tid] = m; ri[tid] = mi;
    __syncthreads();
    for (int s = 128; s > 0; s >>= 1) {
      if (tid < s && rv[tid + s] > rv[tid]) { rv[tid] = rv[tid + s]; ri[tid] = ri[tid + s]; }
      __syncthreads();
    }
    if (tid == 0) {
      chart[(((long)i * N_ + j) * B_ + b) * S_ + a] = rv[0];
      int xy = ri[0];
      int x = xy / S_, y = xy % S_;
      int u = argU[(long)bid * SS_ + xy];
      bp[(((long)i * N_ + j) * B_ + b) * NT_ + a] = (x << 16) | (y << 8) | u;
    }
    __syncthreads();
  }
}

__global__ void convF16(const float* __restrict__ src, h16* __restrict__ dst, long n) {
  long i = (long)blockIdx.x * 256 + threadIdx.x;
  if (i < n) dst[i] = (h16)src[i];
}

__global__ void fill_f32(float* __restrict__ p, long n, float v) {
  long i = (long)blockIdx.x * 256 + threadIdx.x;
  if (i < n) p[i] = v;
}

__global__ void add_convert(const float* __restrict__ x, const float* __restrict__ h,
                            float* __restrict__ of, h16* __restrict__ oh, long n) {
  long i = (long)blockIdx.x * 256 + threadIdx.x;
  if (i < n) { float v = x[i] + h[i]; of[i] = v; oh[i] = (h16)v; }
}

__global__ void embed_gather(const float* __restrict__ emb, const int* __restrict__ captions,
                             h16* __restrict__ dst) {
  long idx = (long)blockIdx.x * 256 + threadIdx.x;
  if (idx >= (long)B_ * N_ * WD_) return;
  int k = (int)(idx % WD_);
  int r = (int)(idx / WD_);
  int wid = captions[r];
  dst[idx] = (h16)emb[(long)wid * WD_ + k];
}

__global__ void lstm_cell(const float* __restrict__ xproj, const float* __restrict__ gatesh,
                          const float* __restrict__ bias_f, const float* __restrict__ bias_b,
                          float* __restrict__ cstate, float* __restrict__ hs,
                          h16* __restrict__ hprev16, int step) {
  const int j = blockIdx.x * 256 + threadIdx.x;
  const int b = blockIdx.y;
  const int d = blockIdx.z;
  const int t = d ? (N_ - 1 - step) : step;
  const float* gx = xproj + ((long)d * (B_ * N_) + b * N_ + t) * (4 * HD_);
  const float* gh = gatesh + ((long)d * B_ + b) * (4 * HD_);
  const float* bias = d ? bias_b : bias_f;
  float gi = gx[j] + gh[j] + bias[j];
  float gf = gx[HD_ + j] + gh[HD_ + j] + bias[HD_ + j];
  float gg = gx[2 * HD_ + j] + gh[2 * HD_ + j] + bias[2 * HD_ + j];
  float go = gx[3 * HD_ + j] + gh[3 * HD_ + j] + bias[3 * HD_ + j];
  float* cp = cstate + ((long)d * B_ + b) * HD_ + j;
  float c = sigm(gf) * (*cp) + sigm(gi) * tanhf(gg);
  *cp = c;
  float h = sigm(go) * tanhf(c);
  hs[(((long)d * N_ + t) * B_ + b) * HD_ + j] = h;
  hprev16[((long)d * B_ + b) * HD_ + j] = (h16)h;
}

__global__ void pool_max(const float* __restrict__ hs, float* __restrict__ hmax32,
                         h16* __restrict__ hmax16) {
  int idx = blockIdx.x * 256 + threadIdx.x;
  if (idx >= B_ * 2 * HD_) return;
  int b = idx / (2 * HD_);
  int jc = idx % (2 * HD_);
  int d = jc / HD_, jj = jc % HD_;
  float m = -3.0e38f;
  for (int t = 0; t < N_; ++t) m = fmaxf(m, hs[(((long)d * N_ + t) * B_ + b) * HD_ + jj]);
  hmax32[(long)b * (2 * HD_) + jc] = m;
  hmax16[(long)b * (2 * HD_) + jc] = (h16)m;
}

__global__ void kl_kernel(const float* __restrict__ enc, float* __restrict__ klb) {
  int b = threadIdx.x;
  if (b >= B_) return;
  float acc = 0.f;
  for (int d = 0; d < ZD_; ++d) {
    float mean = enc[b * 128 + d];
    float lv = enc[b * 128 + ZD_ + d];
    acc += mean * mean + expf(lv) - lv - 1.f;
  }
  klb[b] = 0.5f * acc;
}

__global__ void concat_ez(const float* __restrict__ embTab, const float* __restrict__ enc,
                          h16* __restrict__ dst, int nEmb) {
  long idx = (long)blockIdx.x * 256 + threadIdx.x;
  long total = (long)B_ * nEmb * (SD_ + ZD_);
  if (idx >= total) return;
  int k = (int)(idx % (SD_ + ZD_));
  long r = idx / (SD_ + ZD_);
  int b = (int)(r / nEmb);
  int e = (int)(r % nEmb);
  float v = (k < SD_) ? embTab[(long)e * SD_ + k] : enc[b * 128 + (k - SD_)];
  dst[idx] = (h16)v;
}

__global__ void roots_lsm(const float* __restrict__ logits, float* __restrict__ roots) {
  int b = threadIdx.x;
  if (b >= B_) return;
  const float* x = logits + b * 16;
  float m = -3.0e38f;
  for (int a = 0; a < NT_; ++a) m = fmaxf(m, x[a]);
  float s = 0.f;
  for (int a = 0; a < NT_; ++a) s += expf(x[a] - m);
  float l = m + logf(s);
  for (int a = 0; a < NT_; ++a) roots[b * 16 + a] = x[a] - l;
}

__global__ void row_lse(const float* __restrict__ X, int C, float* __restrict__ lse,
                        float* __restrict__ rowmax) {
  const int row = blockIdx.x;
  const float* x = X + (long)row * C;
  __shared__ float red[256];
  __shared__ float sm;
  int tid = threadIdx.x;
  float m = -3.0e38f;
  for (int c = tid; c < C; c += 256) m = fmaxf(m, x[c]);
  red[tid] = m; __syncthreads();
  for (int s = 128; s > 0; s >>= 1) { if (tid < s) red[tid] = fmaxf(red[tid], red[tid + s]); __syncthreads(); }
  if (tid == 0) sm = red[0];
  __syncthreads();
  float mm = sm;
  float sum = 0.f;
  for (int c = tid; c < C; c += 256) sum += expf(x[c] - mm);
  red[tid] = sum; __syncthreads();
  for (int s = 128; s > 0; s >>= 1) { if (tid < s) red[tid] += red[tid + s]; __syncthreads(); }
  if (tid == 0) {
    lse[row] = mm + logf(red[0]);
    if (rowmax) rowmax[row] = mm;
  }
}

__global__ void terms_gather(const float* __restrict__ logits, const float* __restrict__ lse,
                             const int* __restrict__ captions, float* __restrict__ terms) {
  int idx = blockIdx.x * 256 + threadIdx.x;
  if (idx >= B_ * N_ * T_) return;
  int tt = idx % T_;
  int n = (idx / T_) % N_;
  int b = idx / (T_ * N_);
  int row = b * T_ + tt;
  int wid = captions[b * N_ + n];
  terms[((long)b * N_ + n) * T_ + tt] = logits[(long)row * V_ + wid] - lse[row];
}

__global__ void rules_build(const float* __restrict__ logits, const float* __restrict__ lse,
                            const float* __restrict__ rowmax, float* __restrict__ rules,
                            h16* __restrict__ Rexp, float* __restrict__ rmaxv) {
  long idx = (long)blockIdx.x * 256 + threadIdx.x;
  if (idx >= (long)B_ * NT_ * SS_) return;
  int xy = (int)(idx % SS_);
  int row = (int)(idx / SS_);
  int b = row / NT_, a = row % NT_;
  float v = logits[idx];
  rules[idx] = v - lse[row];
  Rexp[((long)b * 16 + a) * 2048 + xy] = (h16)expf(v - rowmax[row]);
  if (xy == 0) rmaxv[row] = rowmax[row] - lse[row];
}

__global__ void diag_init(const float* __restrict__ terms, float* __restrict__ chartZ,
                          float* __restrict__ chartV) {
  int idx = blockIdx.x * 256 + threadIdx.x;
  if (idx >= N_ * B_ * T_) return;
  int tt = idx % T_;
  int b = (idx / T_) % B_;
  int n = idx / (T_ * B_);
  float v = terms[((long)b * N_ + n) * T_ + tt];
  long pos = (((long)n * N_ + n) * B_ + b) * S_ + NT_ + tt;
  chartZ[pos] = v;
  chartV[pos] = v;
}

__global__ void logz_root(const float* __restrict__ chart, const float* __restrict__ roots,
                          const int* __restrict__ lengths, float* __restrict__ nll) {
  int b = threadIdx.x;
  if (b >= B_) return;
  int len = lengths[b];
  const float* base = chart + (((long)(len - 1)) * B_ + b) * S_;
  float m = -3.0e38f;
  for (int a = 0; a < NT_; ++a) m = fmaxf(m, base[a] + roots[b * 16 + a]);
  float s = 0.f;
  for (int a = 0; a < NT_; ++a) s += expf(base[a] + roots[b * 16 + a] - m);
  nll[b] = -(m + logf(s));
}

__global__ void backtrace(const float* __restrict__ chart, const int* __restrict__ bp,
                          const float* __restrict__ roots, const int* __restrict__ lengths,
                          float* __restrict__ out) {
  int b = threadIdx.x;
  if (b >= B_) return;
  int len = lengths[b];
  int j0 = len - 1;
  float m = -3.0e38f; int ba = 0;
  for (int a = 0; a < NT_; ++a) {
    float v = chart[(((long)j0) * B_ + b) * S_ + a] + roots[b * 16 + a];
    if (v > m) { m = v; ba = a; }
  }
  int sti[64], stj[64], sta[64];
  int sp = 0;
  if (j0 > 0) { sti[0] = 0; stj[0] = j0; sta[0] = ba; sp = 1; }
  while (sp > 0) {
    --sp;
    int i = sti[sp], j = stj[sp], aa = sta[sp];
    out[((long)b * N_ + i) * N_ + j] = 1.0f;
    int v = bp[(((long)i * N_ + j) * B_ + b) * NT_ + aa];
    int x = (v >> 16) & 255, y = (v >> 8) & 255, u = v & 255;
    int k = i + u;
    if (k > i)     { sti[sp] = i;     stj[sp] = k; sta[sp] = x; ++sp; }
    if (j > k + 1) { sti[sp] = k + 1; stj[sp] = j; sta[sp] = y; ++sp; }
  }
}

__global__ void finalize(const float* __restrict__ nll, const float* __restrict__ klb,
                         const int* __restrict__ lengths, float* __restrict__ out) {
  int b = threadIdx.x;
  __shared__ float sn[B_], sk[B_];
  __shared__ int sl2[B_];
  if (b < B_) {
    out[B_ * N_ * N_ + b] = nll[b] + klb[b];  // VSE_LM_ALPHA == 1
    sn[b] = nll[b]; sk[b] = klb[b]; sl2[b] = lengths[b];
  }
  __syncthreads();
  if (b == 0) {
    float a = 0.f, k = 0.f; int L = 0;
    for (int x = 0; x < B_; ++x) { a += sn[x]; k += sk[x]; L += sl2[x]; }
    float denom = (float)L;
    out[B_ * N_ * N_ + B_ + 0] = a / denom;
    out[B_ * N_ * N_ + B_ + 1] = k / (float)B_;
    out[B_ * N_ * N_ + B_ + 2] = (a + k) / denom;
  }
}

// -------------------------- host side --------------------------

static void gemm(hipStream_t s, const h16* A, int lda, long sA, const h16* Bt, int ldb, long sB,
                 const float* bias, float* Cf, int ldcf, long sCf, h16* Ch, int ldch, long sCh,
                 int M, int N, int K, int act, int batches) {
  dim3 g((N + 31) / 32, (M + 31) / 32, batches);
  wmma_gemm<<<g, 32, 0, s>>>(A, lda, sA, Bt, ldb, sB, bias, Cf, ldcf, sCf, Ch, ldch, sCh, M, N, K, act);
}

#define CDIV(a, b) (((a) + (b) - 1) / (b))

extern "C" void kernel_launch(void* const* d_in, const int* in_sizes, int n_in,
                              void* d_out, int out_size, void* d_ws, size_t ws_size,
                              hipStream_t stream) {
  (void)in_sizes; (void)n_in; (void)ws_size;
  // ---- inputs in setup_inputs() dict insertion order, recursively flattened ----
  const float* enc_emb   = (const float*)d_in[0];
  const float* wih_f     = (const float*)d_in[1];
  const float* whh_f     = (const float*)d_in[2];
  const float* b_f       = (const float*)d_in[3];
  const float* wih_b     = (const float*)d_in[4];
  const float* whh_b     = (const float*)d_in[5];
  const float* b_b       = (const float*)d_in[6];
  const float* encw      = (const float*)d_in[7];
  const float* encb      = (const float*)d_in[8];
  const float* term_emb  = (const float*)d_in[9];
  const float* nt_emb    = (const float*)d_in[10];
  const float* root_emb  = (const float*)d_in[11];
  const float* rulew     = (const float*)d_in[12];
  const float* ruleb     = (const float*)d_in[13];
  const float* r_li_w    = (const float*)d_in[14];
  const float* r_li_b    = (const float*)d_in[15];
  const float* r_11_w    = (const float*)d_in[16];
  const float* r_11_b    = (const float*)d_in[17];
  const float* r_12_w    = (const float*)d_in[18];
  const float* r_12_b    = (const float*)d_in[19];
  const float* r_21_w    = (const float*)d_in[20];
  const float* r_21_b    = (const float*)d_in[21];
  const float* r_22_w    = (const float*)d_in[22];
  const float* r_22_b    = (const float*)d_in[23];
  const float* r_lo_w    = (const float*)d_in[24];
  const float* r_lo_b    = (const float*)d_in[25];
  const float* t_li_w    = (const float*)d_in[26];
  const float* t_li_b    = (const float*)d_in[27];
  const float* t_11_w    = (const float*)d_in[28];
  const float* t_11_b    = (const float*)d_in[29];
  const float* t_12_w    = (const float*)d_in[30];
  const float* t_12_b    = (const float*)d_in[31];
  const float* t_21_w    = (const float*)d_in[32];
  const float* t_21_b    = (const float*)d_in[33];
  const float* t_22_w    = (const float*)d_in[34];
  const float* t_22_b    = (const float*)d_in[35];
  const float* t_lo_w    = (const float*)d_in[36];
  const float* t_lo_b    = (const float*)d_in[37];
  const int*   captions  = (const int*)d_in[38];
  const int*   lengths   = (const int*)d_in[39];
  float* out = (float*)d_out;

  // ---- workspace bump allocator (256B aligned) ----
  size_t off = 0;
  auto alloc = [&](size_t bytes) -> void* {
    off = (off + 255) & ~(size_t)255;
    void* p = (char*)d_ws + off;
    off += bytes;
    return p;
  };
  h16*   emb16    = (h16*)alloc((size_t)B_ * N_ * WD_ * 2);
  h16*   wih16    = (h16*)alloc((size_t)2 * 4 * HD_ * WD_ * 2);
  h16*   whh16    = (h16*)alloc((size_t)2 * 4 * HD_ * HD_ * 2);
  float* xproj    = (float*)alloc((size_t)2 * B_ * N_ * 4 * HD_ * 4);
  float* gatesh   = (float*)alloc((size_t)2 * B_ * 4 * HD_ * 4);
  float* cstate   = (float*)alloc((size_t)2 * B_ * HD_ * 4);
  h16*   hprev16  = (h16*)alloc((size_t)2 * B_ * HD_ * 2);
  float* hs       = (float*)alloc((size_t)2 * N_ * B_ * HD_ * 4);
  float* hmax32   = (float*)alloc((size_t)B_ * 2 * HD_ * 4);
  h16*   hmax16   = (h16*)alloc((size_t)B_ * 2 * HD_ * 2);
  h16*   encw16   = (h16*)alloc((size_t)128 * 1024 * 2);
  float* encout   = (float*)alloc((size_t)B_ * 128 * 4);
  float* klb      = (float*)alloc((size_t)B_ * 4);
  h16*   root_in16 = (h16*)alloc((size_t)B_ * 320 * 2);
  h16*   term_in16 = (h16*)alloc((size_t)B_ * T_ * 320 * 2);
  h16*   nt_in16   = (h16*)alloc((size_t)B_ * NT_ * 320 * 2);
  h16*   rli16 = (h16*)alloc((size_t)256 * 320 * 2);
  h16*   r1116 = (h16*)alloc((size_t)256 * 256 * 2);
  h16*   r1216 = (h16*)alloc((size_t)256 * 256 * 2);
  h16*   r2116 = (h16*)alloc((size_t)256 * 256 * 2);
  h16*   r2216 = (h16*)alloc((size_t)256 * 256 * 2);
  h16*   rlo16 = (h16*)alloc((size_t)32 * 256 * 2);      // 15 rows padded to 32
  h16*   tli16 = (h16*)alloc((size_t)256 * 320 * 2);
  h16*   t1116 = (h16*)alloc((size_t)256 * 256 * 2);
  h16*   t1216 = (h16*)alloc((size_t)256 * 256 * 2);
  h16*   t2116 = (h16*)alloc((size_t)256 * 256 * 2);
  h16*   t2216 = (h16*)alloc((size_t)256 * 256 * 2);
  h16*   tlo16 = (h16*)alloc((size_t)10016 * 256 * 2);   // 10000 rows padded to 10016
  h16*   rulew16 = (h16*)alloc((size_t)2048 * 320 * 2);  // 2025 rows padded to 2048
  h16*   h16a  = (h16*)alloc((size_t)960 * 256 * 2);
  h16*   h16b  = (h16*)alloc((size_t)960 * 256 * 2);
  float* hf32a = (float*)alloc((size_t)960 * 256 * 4);
  float* hf32b = (float*)alloc((size_t)960 * 256 * 4);
  float* root_logits = (float*)alloc((size_t)B_ * 16 * 4);
  float* roots       = (float*)alloc((size_t)B_ * 16 * 4);
  float* term_logits = (float*)alloc((size_t)B_ * T_ * V_ * 4);
  float* lse_t       = (float*)alloc((size_t)B_ * T_ * 4);
  float* terms       = (float*)alloc((size_t)B_ * N_ * T_ * 4);
  float* rule_logits = (float*)alloc((size_t)B_ * NT_ * SS_ * 4);
  float* rule_lse    = (float*)alloc((size_t)B_ * NT_ * 4);
  float* rule_rm     = (float*)alloc((size_t)B_ * NT_ * 4);
  float* rmaxv       = (float*)alloc((size_t)B_ * NT_ * 4);
  float* rules       = (float*)alloc((size_t)B_ * NT_ * SS_ * 4);
  h16*   Rexp16      = (h16*)alloc((size_t)B_ * 16 * 2048 * 2);
  float* chartZ      = (float*)alloc((size_t)N_ * N_ * B_ * S_ * 4);
  float* chartV      = (float*)alloc((size_t)N_ * N_ * B_ * S_ * 4);
  float* cmbuf       = (float*)alloc((size_t)23 * B_ * 4);
  h16*   F16         = (h16*)alloc((size_t)B_ * 32 * 2048 * 2);
  float* Gbuf        = (float*)alloc((size_t)23 * B_ * SS_ * 4);
  unsigned char* argU = (unsigned char*)alloc((size_t)23 * B_ * SS_);
  int*   bpbuf       = (int*)alloc((size_t)N_ * N_ * B_ * NT_ * 4);
  float* nll         = (float*)alloc((size_t)B_ * 4);

  // zero workspace (pads must be zero for WMMA padding) + zero output
  hipMemsetAsync(d_ws, 0, off, stream);
  hipMemsetAsync(d_out, 0, (size_t)out_size * 4, stream);

  auto conv = [&](const float* src, h16* dst, long n) {
    convF16<<<dim3((unsigned)CDIV(n, 256)), 256, 0, stream>>>(src, dst, n);
  };
  // weight conversions to f16
  conv(wih_f, wih16, (long)4 * HD_ * WD_);
  conv(wih_b, wih16 + (long)4 * HD_ * WD_, (long)4 * HD_ * WD_);
  conv(whh_f, whh16, (long)4 * HD_ * HD_);
  conv(whh_b, whh16 + (long)4 * HD_ * HD_, (long)4 * HD_ * HD_);
  conv(encw, encw16, (long)128 * 1024);
  conv(rulew, rulew16, (long)SS_ * 320);
  conv(r_li_w, rli16, 256 * 320); conv(r_11_w, r1116, 256 * 256); conv(r_12_w, r1216, 256 * 256);
  conv(r_21_w, r2116, 256 * 256); conv(r_22_w, r2216, 256 * 256); conv(r_lo_w, rlo16, NT_ * 256);
  conv(t_li_w, tli16, 256 * 320); conv(t_11_w, t1116, 256 * 256); conv(t_12_w, t1216, 256 * 256);
  conv(t_21_w, t2116, 256 * 256); conv(t_22_w, t2216, 256 * 256); conv(t_lo_w, tlo16, (long)V_ * 256);

  // encoder
  embed_gather<<<dim3(CDIV(B_ * N_ * WD_, 256)), 256, 0, stream>>>(enc_emb, captions, emb16);
  // x @ W_ih^T for both directions (strideA = 0 shares embeddings)
  gemm(stream, emb16, WD_, 0, wih16, WD_, (long)4 * HD_ * WD_, nullptr,
       xproj, 4 * HD_, (long)B_ * N_ * 4 * HD_, nullptr, 0, 0,
       B_ * N_, 4 * HD_, WD_, 0, 2);
  for (int step = 0; step < N_; ++step) {
    gemm(stream, hprev16, HD_, (long)B_ * HD_, whh16, HD_, (long)4 * HD_ * HD_, nullptr,
         gatesh, 4 * HD_, (long)B_ * 4 * HD_, nullptr, 0, 0,
         B_, 4 * HD_, HD_, 0, 2);
    lstm_cell<<<dim3(2, B_, 2), 256, 0, stream>>>(xproj, gatesh, b_f, b_b, cstate, hs, hprev16, step);
  }
  pool_max<<<dim3(CDIV(B_ * 2 * HD_, 256)), 256, 0, stream>>>(hs, hmax32, hmax16);
  gemm(stream, hmax16, 2 * HD_, 0, encw16, 2 * HD_, 0, encb,
       encout, 128, 0, nullptr, 0, 0, B_, 128, 2 * HD_, 0, 1);
  kl_kernel<<<1, 32, 0, stream>>>(encout, klb);

  // concat inputs [emb | z]
  concat_ez<<<dim3(CDIV(B_ * 1 * 320, 256)), 256, 0, stream>>>(root_emb, encout, root_in16, 1);
  concat_ez<<<dim3(CDIV(B_ * T_ * 320, 256)), 256, 0, stream>>>(term_emb, encout, term_in16, T_);
  concat_ez<<<dim3(CDIV(B_ * NT_ * 320, 256)), 256, 0, stream>>>(nt_emb, encout, nt_in16, NT_);

  // ---- root MLP (M = 32) ----
  {
    const int M = B_;
    gemm(stream, root_in16, 320, 0, rli16, 320, 0, r_li_b, hf32a, 256, 0, h16a, 256, 0, M, 256, 320, 0, 1);
    gemm(stream, h16a, 256, 0, r1116, 256, 0, r_11_b, nullptr, 0, 0, h16b, 256, 0, M, 256, 256, 1, 1);
    gemm(stream, h16b, 256, 0, r1216, 256, 0, r_12_b, hf32b, 256, 0, nullptr, 0, 0, M, 256, 256, 1, 1);
    add_convert<<<dim3(CDIV(M * 256, 256)), 256, 0, stream>>>(hf32a, hf32b, hf32a, h16a, (long)M * 256);
    gemm(stream, h16a, 256, 0, r2116, 256, 0, r_21_b, nullptr, 0, 0, h16b, 256, 0, M, 256, 256, 1, 1);
    gemm(stream, h16b, 256, 0, r2216, 256, 0, r_22_b, hf32b, 256, 0, nullptr, 0, 0, M, 256, 256, 1, 1);
    add_convert<<<dim3(CDIV(M * 256, 256)), 256, 0, stream>>>(hf32a, hf32b, hf32a, h16a, (long)M * 256);
    gemm(stream, h16a, 256, 0, rlo16, 256, 0, r_lo_b, root_logits, 16, 0, nullptr, 0, 0, M, NT_, 256, 0, 1);
    roots_lsm<<<1, 32, 0, stream>>>(root_logits, roots);
  }

  // ---- term MLP (M = 960) ----
  {
    const int M = B_ * T_;
    gemm(stream, term_in16, 320, 0, tli16, 320, 0, t_li_b, hf32a, 256, 0, h16a, 256, 0, M, 256, 320, 0, 1);
    gemm(stream, h16a, 256, 0, t1116, 256, 0, t_11_b, nullptr, 0, 0, h16b, 256, 0, M, 256, 256, 1, 1);
    gemm(stream, h16b, 256, 0, t1216, 256, 0, t_12_b, hf32b, 256, 0, nullptr, 0, 0, M, 256, 256, 1, 1);
    add_convert<<<dim3(CDIV(M * 256, 256)), 256, 0, stream>>>(hf32a, hf32b, hf32a, h16a, (long)M * 256);
    gemm(stream, h16a, 256, 0, t2116, 256, 0, t_21_b, nullptr, 0, 0, h16b, 256, 0, M, 256, 256, 1, 1);
    gemm(stream, h16b, 256, 0, t2216, 256, 0, t_22_b, hf32b, 256, 0, nullptr, 0, 0, M, 256, 256, 1, 1);
    add_convert<<<dim3(CDIV(M * 256, 256)), 256, 0, stream>>>(hf32a, hf32b, hf32a, h16a, (long)M * 256);
    gemm(stream, h16a, 256, 0, tlo16, 256, 0, t_lo_b, term_logits, V_, 0, nullptr, 0, 0, M, V_, 256, 0, 1);
    row_lse<<<dim3(M), 256, 0, stream>>>(term_logits, V_, lse_t, nullptr);
    terms_gather<<<dim3(CDIV(B_ * N_ * T_, 256)), 256, 0, stream>>>(term_logits, lse_t, captions, terms);
  }

  // ---- rule head (M = 480, N = 2025) ----
  gemm(stream, nt_in16, 320, 0, rulew16, 320, 0, ruleb, rule_logits, SS_, 0, nullptr, 0, 0,
       B_ * NT_, SS_, 320, 0, 1);
  row_lse<<<dim3(B_ * NT_), 256, 0, stream>>>(rule_logits, SS_, rule_lse, rule_rm);
  rules_build<<<dim3(CDIV(B_ * NT_ * SS_, 256)), 256, 0, stream>>>(rule_logits, rule_lse, rule_rm,
                                                                   rules, Rexp16, rmaxv);

  // ---- charts ----
  long chartN = (long)N_ * N_ * B_ * S_;
  fill_f32<<<dim3((unsigned)CDIV(chartN, 256)), 256, 0, stream>>>(chartZ, chartN, NEGV);
  fill_f32<<<dim3((unsigned)CDIV(chartN, 256)), 256, 0, stream>>>(chartV, chartN, NEGV);
  diag_init<<<dim3(CDIV(N_ * B_ * T_, 256)), 256, 0, stream>>>(terms, chartZ, chartV);

  for (int w = 2; w <= N_; ++w) {
    int ni = N_ - w + 1;
    logz_combine<<<dim3(ni * B_), 256, 0, stream>>>(chartZ, F16, cmbuf, w);
    inside_gemm<<<dim3(1, B_), 32, 0, stream>>>(F16, Rexp16, cmbuf, rmaxv, chartZ, w, ni);
    vit_combine<<<dim3(ni * B_), 256, 0, stream>>>(chartV, Gbuf, argU, w);
    vit_reduce<<<dim3(ni * B_), 256, 0, stream>>>(Gbuf, argU, rules, chartV, bpbuf, w);
  }

  logz_root<<<1, 32, 0, stream>>>(chartZ, roots, lengths, nll);
  backtrace<<<1, 32, 0, stream>>>(chartV, bpbuf, roots, lengths, out);
  finalize<<<1, 32, 0, stream>>>(nll, klb, lengths, out);
}
